// KGATRecommender_1039382086191
// MI455X (gfx1250) — compile-verified, matrix operand order checked by reading, not run on hardware
//
#include <hip/hip_runtime.h>
#include <hip/hip_bf16.h>
#include <math.h>

// ---------------------------------------------------------------------------
// KGAT forward for MI455X (gfx1250, wave32, WMMA).
//
// Per layer:
//   1) edge kernel: wave-per-edge attention logit + sigmoid gate, scatter-add
//      gated src embedding into agg[] with global_atomic_add_f32.
//   2) WMMA GEMM: x_next = relu([x | agg/deg] @ W(256x128) + b) using
//      v_wmma_f32_16x16x32_f16. Weights are converted f32->f16 once per block
//      into LDS *pre-swizzled in B-fragment order*, so each fragment is one
//      aligned 32-byte LDS read (2x ds_load_b128) instead of 16x ds_load_u16.
// ---------------------------------------------------------------------------

#define DIM      128
#define TWO_DIM  256

typedef __attribute__((ext_vector_type(16))) _Float16 v16h;
typedef __attribute__((ext_vector_type(8)))  float    v8f;

// ------------------------- utility kernels ---------------------------------

__global__ void zero_f32_kernel(float* __restrict__ p, size_t n) {
    size_t i = (size_t)blockIdx.x * blockDim.x + threadIdx.x;
    size_t stride = (size_t)gridDim.x * blockDim.x;
    for (; i < n; i += stride) p[i] = 0.0f;
}

// degree of dst nodes: one thread per edge
__global__ void degree_kernel(const int* __restrict__ dst, float* __restrict__ deg, int E) {
    int i = blockIdx.x * blockDim.x + threadIdx.x;
    int stride = gridDim.x * blockDim.x;
    for (; i < E; i += stride) {
        atomicAdd(&deg[dst[i]], 1.0f);
    }
}

// ------------------------- edge attention + scatter ------------------------
// One wave (32 lanes) per edge. Each lane handles 4 consecutive features
// (float4), so 32 lanes cover DIM=128. Logit reduced with shfl_xor tree.
__global__ void edge_kernel(const float* __restrict__ x,
                            const int* __restrict__ src,
                            const int* __restrict__ dst,
                            const float* __restrict__ att_w,   // [256]
                            const float* __restrict__ att_b,   // [1]
                            float* __restrict__ agg,           // [N,128]
                            int E) {
    int lane = threadIdx.x & 31;
    int edge = (blockIdx.x * blockDim.x + threadIdx.x) >> 5;
    if (edge >= E) return;

    int s = src[edge];
    int d = dst[edge];

    const float4 sv = ((const float4*)(x + (size_t)s * DIM))[lane];
    const float4 dv = ((const float4*)(x + (size_t)d * DIM))[lane];
    const float4 ws = ((const float4*)att_w)[lane];
    const float4 wd = ((const float4*)(att_w + DIM))[lane];

    float p = sv.x * ws.x + sv.y * ws.y + sv.z * ws.z + sv.w * ws.w
            + dv.x * wd.x + dv.y * wd.y + dv.z * wd.z + dv.w * wd.w;

    // wave32 tree reduction
    #pragma unroll
    for (int o = 16; o >= 1; o >>= 1) p += __shfl_xor(p, o, 32);

    float logit = p + att_b[0];
    float g = 1.0f / (1.0f + __expf(-logit));

    float* out = agg + (size_t)d * DIM + lane * 4;
    atomicAdd(out + 0, sv.x * g);
    atomicAdd(out + 1, sv.y * g);
    atomicAdd(out + 2, sv.z * g);
    atomicAdd(out + 3, sv.w * g);
}

// ------------------------- WMMA GEMM ---------------------------------------
// x_out[row, :] = relu( concat(x[row,:], agg[row,:]/deg[row]) @ W + bias )
// W is [256, 128] f32 in global. Converted once per block into LDS laid out
// in B-fragment order:
//   Wswz[(((kc*8 + t)*32 + lane)*16) + e] = f16( W[k*128 + n] )
// with  k = kc*32 + (e&7) + (e>>3)*16 + (lane>>4)*8,  n = t*16 + (lane&15).
// Fragment load = one aligned v16h LDS read.
//
// Block = 256 threads = 8 waves. Each wave: one 16-row tile, full 128 cols.
// K = 256 processed in 8 chunks of 32; 8 column tiles of 16.
__global__ void __launch_bounds__(256)
gemm_wmma_kernel(const float* __restrict__ xin,   // [N,128]
                 const float* __restrict__ agg,   // [N,128]
                 const float* __restrict__ deg,   // [N]
                 const float* __restrict__ W,     // [256,128]
                 const float* __restrict__ bias,  // [128]
                 float* __restrict__ xout,        // [N,128]
                 int Nrows) {
    // 8 kc * 8 t * 32 lanes fragments, 16 halves each = 64 KB, 32B aligned
    __shared__ __align__(32) _Float16 Wswz[8 * 8 * 32 * 16];

    // cooperative f32 -> f16 conversion into swizzled fragment layout
    for (int i = threadIdx.x; i < 8 * 8 * 32 * 16; i += blockDim.x) {
        const int e  = i & 15;
        const int ln = (i >> 4) & 31;
        const int t  = (i >> 9) & 7;
        const int kc = i >> 12;
        const int k  = kc * 32 + (e & 7) + ((e >> 3) * 16) + ((ln >> 4) * 8);
        const int n  = t * 16 + (ln & 15);
        Wswz[i] = (_Float16)W[k * DIM + n];
    }
    __syncthreads();

    const int wave = threadIdx.x >> 5;
    const int lane = threadIdx.x & 31;
    const int rowbase = (blockIdx.x * 8 + wave) * 16;
    if (rowbase >= Nrows) return;            // wave-uniform: EXEC stays all-1s

    // A-operand: lane L holds row rowbase+(L&15); its K slots follow
    //   k(e) = (e&7) + (e>>3)*16 + (L>>4)*8   within each 32-wide chunk.
    const int arow   = rowbase + (lane & 15);
    const int arowc  = arow < Nrows ? arow : Nrows - 1;   // clamp (safe loads)
    const int klane  = (lane >> 4) * 8;
    const float* xrow = xin + (size_t)arowc * DIM;
    const float* grow = agg + (size_t)arowc * DIM;
    const float invdeg = 1.0f / fmaxf(deg[arowc], 1.0f);

    // B fragments for this lane, in swizzled order
    const v16h* bfrag = (const v16h*)Wswz + lane;   // index: (kc*8 + t)*32 + lane

    // C/D mapping: lane gives column (lane&15), VGPR r gives row r+(lane>>4)*8
    const int ncol = lane & 15;
    const int mrow = (lane >> 4) * 8;

    v8f acc[8];
    #pragma unroll
    for (int t = 0; t < 8; ++t) acc[t] = (v8f){0.f,0.f,0.f,0.f,0.f,0.f,0.f,0.f};

    #pragma unroll
    for (int kc = 0; kc < 8; ++kc) {         // K chunks of 32 (total K=256)
        const int kbase = kc * 32;

        // build A fragment (f32 -> f16 on the fly; k>=128 comes from agg/deg)
        // K slots are contiguous in runs of 8 -> compiler emits b128 loads.
        v16h a;
        #pragma unroll
        for (int e = 0; e < 16; ++e) {
            const int k = kbase + (e & 7) + ((e >> 3) * 16) + klane;
            float v = (k < DIM) ? xrow[k] : grow[k - DIM] * invdeg;
            a[e] = (_Float16)v;
        }

        #pragma unroll
        for (int t = 0; t < 8; ++t) {        // 8 column tiles of 16
            const v16h b = bfrag[(kc * 8 + t) * 32];
            acc[t] = __builtin_amdgcn_wmma_f32_16x16x32_f16(
                /*neg_a=*/false, a, /*neg_b=*/false, b,
                /*c_mod=*/(short)0, acc[t],
                /*reuse_a=*/false, /*reuse_b=*/false);
        }
    }

    // epilogue: bias + relu, store (lanes 0-15 cover 16 consecutive cols)
    #pragma unroll
    for (int t = 0; t < 8; ++t) {
        const int col = t * 16 + ncol;
        const float bcol = bias[col];
        #pragma unroll
        for (int r = 0; r < 8; ++r) {
            const int row = rowbase + mrow + r;
            if (row < Nrows) {
                float v = acc[t][r] + bcol;
                xout[(size_t)row * DIM + col] = v > 0.0f ? v : 0.0f;
            }
        }
    }
}

// ------------------------- launch ------------------------------------------

extern "C" void kernel_launch(void* const* d_in, const int* in_sizes, int n_in,
                              void* d_out, int out_size, void* d_ws, size_t ws_size,
                              hipStream_t stream) {
    const int*   edge_index = (const int*)  d_in[0];  // [2, E]
    const float* user_emb   = (const float*)d_in[1];  // [NU, 128]
    const float* item_emb   = (const float*)d_in[2];  // [NI, 128]
    const float* att_w      = (const float*)d_in[3];  // [3, 256, 1]
    const float* att_b      = (const float*)d_in[4];  // [3, 1]
    const float* agg_w      = (const float*)d_in[5];  // [3, 256, 128]
    const float* agg_b      = (const float*)d_in[6];  // [3, 128]

    const int E  = in_sizes[0] / 2;
    const int NU = in_sizes[1] / DIM;
    const int NI = in_sizes[2] / DIM;
    const int N  = NU + NI;

    const int* src = edge_index;
    const int* dst = edge_index + E;

    // workspace layout: x0 | x1 | agg | deg
    float* x0  = (float*)d_ws;
    float* x1  = x0 + (size_t)N * DIM;
    float* agg = x1 + (size_t)N * DIM;
    float* deg = agg + (size_t)N * DIM;

    const size_t ND = (size_t)N * DIM;

    // x0 = concat(user_emb, item_emb)
    hipMemcpyAsync(x0, user_emb, (size_t)NU * DIM * sizeof(float),
                   hipMemcpyDeviceToDevice, stream);
    hipMemcpyAsync(x0 + (size_t)NU * DIM, item_emb, (size_t)NI * DIM * sizeof(float),
                   hipMemcpyDeviceToDevice, stream);

    // degree (constant across layers)
    zero_f32_kernel<<<1024, 256, 0, stream>>>(deg, (size_t)N);
    degree_kernel<<<(E + 255) / 256, 256, 0, stream>>>(dst, deg, E);

    const int edge_blocks = (E * 32 + 255) / 256;         // wave per edge
    const int row_tiles   = (N + 15) / 16;
    const int gemm_blocks = (row_tiles + 7) / 8;          // 8 waves per block

    const float* xcur = x0;
    for (int l = 0; l < 3; ++l) {
        float* xnext = (l == 2) ? (float*)d_out : ((l == 0) ? x1 : x0);

        zero_f32_kernel<<<4096, 256, 0, stream>>>(agg, ND);
        edge_kernel<<<edge_blocks, 256, 0, stream>>>(
            xcur, src, dst, att_w + (size_t)l * TWO_DIM, att_b + l, agg, E);
        gemm_wmma_kernel<<<gemm_blocks, 256, 0, stream>>>(
            xcur, agg, deg, agg_w + (size_t)l * TWO_DIM * DIM,
            agg_b + (size_t)l * DIM, xnext, N);

        xcur = xnext;
    }
    (void)n_in; (void)out_size; (void)ws_size;
}
